// BiMambaWrapper_23398981828919
// MI455X (gfx1250) — matrix operand order repaired and checked by reading, not compiled, not run on hardware
//
#include <hip/hip_runtime.h>
#include <hip/hip_bf16.h>
#include <math.h>

// ---------------------------------------------------------------------------
// Problem dimensions (fixed by the reference's setup_inputs()).
// ---------------------------------------------------------------------------
constexpr int B_  = 1;
constexpr int L_  = 2048;   // sequence length
constexpr int Dm_ = 1024;   // model dim
constexpr int Di_ = 2048;   // inner dim
constexpr int E_  = 2 * Di_;      // 4096, in-proj output cols (x | z)
constexpr int Nst_ = 16;    // SSM state size
constexpr int Kc_ = 4;      // conv kernel
constexpr int R_  = 64;     // dt rank
constexpr int G_  = R_ + 2 * Nst_; // 96, x_dbl cols

typedef __attribute__((ext_vector_type(16))) __bf16 v16bf;
typedef __attribute__((ext_vector_type(8)))  __bf16 v8bf;
typedef __attribute__((ext_vector_type(8)))  float  v8f;

// ---------------------------------------------------------------------------
// f32 -> bf16 conversion
// ---------------------------------------------------------------------------
__global__ void cvt_f32_bf16_kernel(const float* __restrict__ in,
                                    __bf16* __restrict__ out, int n) {
    int i = blockIdx.x * blockDim.x + threadIdx.x;
    if (i < n) out[i] = (__bf16)in[i];
}

// ---------------------------------------------------------------------------
// WMMA GEMM:  C[M,N] (f32) = A[M,K] (bf16) * W[N,K]^T (bf16)
// One wave32 computes a 16 x (16*NT) output tile via v_wmma_f32_16x16x32_bf16.
//
// Fragment layouts used by this code (wave32; A and B share this k order):
//   A 16x32 bf16: lane m = lane&15; khalf = lane>>4;
//       elems 0..15 -> K = khalf*16 + (0..15)
//   B 32x16 bf16: lane n = lane&15; khalf = lane>>4;
//       elems 0..15 -> K = khalf*16 + (0..15)        (contiguous 32B load)
//   C/D f32: VGPR i: row = khalf*8 + i, col = lane&15
//
// EPI: 0 = plain f32 store, 1 = softplus(acc + bias[col])
// ---------------------------------------------------------------------------
template<int NT, int EPI>
__global__ void gemm_wmma_bf16(const __bf16* __restrict__ A,
                               const __bf16* __restrict__ W,
                               const float*  __restrict__ bias,
                               float* __restrict__ C,
                               int M, int N, int K) {
    const int tilesM = M >> 4;
    const int tilesN = N / (16 * NT);
    const int wid = (blockIdx.x * blockDim.x + threadIdx.x) >> 5;
    if (wid >= tilesM * tilesN) return;   // wave-uniform exit: EXEC stays full

    const int mt  = wid % tilesM;
    const int ntg = wid / tilesM;
    const int lane = threadIdx.x & 31;
    const int m16  = lane & 15;
    const int kh   = lane >> 4;

    const __bf16* Arow = A + (size_t)(mt * 16 + m16) * K;
    const int colbase = ntg * (16 * NT);

    v8f acc[NT] = {};

    const int ksteps = K >> 5;
    for (int kk = 0; kk < ksteps; ++kk) {
        const int k0 = kk << 5;
        // A fragment: two 16-byte loads per lane
        v8bf alo = *reinterpret_cast<const v8bf*>(Arow + k0 + kh * 16);
        v8bf ahi = *reinterpret_cast<const v8bf*>(Arow + k0 + kh * 16 + 8);
        v16bf a = __builtin_shufflevector(alo, ahi,
                    0,1,2,3,4,5,6,7,8,9,10,11,12,13,14,15);
#pragma unroll
        for (int j = 0; j < NT; ++j) {
            // B fragment: one 32-byte load per lane (W row-major [N,K])
            const __bf16* Wp = W + (size_t)(colbase + j * 16 + m16) * K
                                 + k0 + kh * 16;
            v16bf b = *reinterpret_cast<const v16bf*>(Wp);
            acc[j] = __builtin_amdgcn_wmma_f32_16x16x32_bf16(
                false, a, false, b, (short)0, acc[j], false, false);
        }
    }

#pragma unroll
    for (int j = 0; j < NT; ++j) {
        const int col = colbase + j * 16 + m16;
#pragma unroll
        for (int i = 0; i < 8; ++i) {
            const int row = mt * 16 + kh * 8 + i;
            float v = acc[j][i];
            if (EPI == 1) {
                v += bias[col];
                v = (v > 20.0f) ? v : log1pf(__expf(v));  // softplus
            }
            C[(size_t)row * N + col] = v;
        }
    }
}

// ---------------------------------------------------------------------------
// Depthwise causal conv (K=4) + bias + SiLU on the x-half of xz.
// The reverse direction reads xz through the flip (xz_rev = flip(xz_fwd)).
// Writes f32 (for scan) and bf16 (for the x@Wx^T GEMM).
// ---------------------------------------------------------------------------
__global__ void conv_silu_kernel(const float* __restrict__ xz,
                                 const float* __restrict__ cw,
                                 const float* __restrict__ cb,
                                 float* __restrict__ xf,
                                 __bf16* __restrict__ xb,
                                 int rev) {
    int idx = blockIdx.x * blockDim.x + threadIdx.x;
    if (idx >= L_ * Di_) return;
    int l = idx / Di_;
    int d = idx % Di_;
    float acc = cb[d];
#pragma unroll
    for (int k = 0; k < Kc_; ++k) {
        int t = l - (Kc_ - 1) + k;
        if (t >= 0) {
            int u = rev ? (L_ - 1 - t) : t;
            acc += cw[d * Kc_ + k] * xz[(size_t)u * E_ + d];
        }
    }
    float s = acc / (1.0f + __expf(-acc));   // silu
    xf[idx] = s;
    xb[idx] = (__bf16)s;
}

// ---------------------------------------------------------------------------
// Extract dt_rank columns of x_dbl as bf16 for the dt GEMM.
// ---------------------------------------------------------------------------
__global__ void extract_dtr_kernel(const float* __restrict__ xdbl,
                                   __bf16* __restrict__ dtr) {
    int idx = blockIdx.x * blockDim.x + threadIdx.x;
    if (idx >= L_ * R_) return;
    int l = idx / R_;
    int r = idx % R_;
    dtr[idx] = (__bf16)xdbl[(size_t)l * G_ + r];
}

// ---------------------------------------------------------------------------
// Selective scan. 16 lanes per channel (one lane per state n), wave32 holds
// two channels. Per step: h_n = exp(dt*A_n)*h_n + dt*x*B_n; y = sum_n h_n*C_n
// via a 4-step shfl_xor tree inside each 16-lane half; lane n==0 applies
// x*D + silu(z) gating and stores.
// ---------------------------------------------------------------------------
__global__ void scan_kernel(const float* __restrict__ dt,
                            const float* __restrict__ xf,
                            const float* __restrict__ xdbl,
                            const float* __restrict__ xz,
                            const float* __restrict__ A_log,
                            const float* __restrict__ Dvec,
                            float* __restrict__ y,
                            int rev) {
    int gtid = blockIdx.x * blockDim.x + threadIdx.x;
    int wave = gtid >> 5;
    int lane = threadIdx.x & 31;
    int n = lane & 15;
    int d = wave * 2 + (lane >> 4);
    if (d >= Di_) return;

    float a  = -__expf(A_log[d * Nst_ + n]);
    float Dp = Dvec[d];
    float s = 0.0f;

    for (int l = 0; l < L_; ++l) {
        float dtv = dt[(size_t)l * Di_ + d];     // uniform within 16-lane group
        float xv  = xf[(size_t)l * Di_ + d];
        float Bn  = xdbl[(size_t)l * G_ + R_ + n];
        float Cn  = xdbl[(size_t)l * G_ + R_ + Nst_ + n];
        s = __expf(dtv * a) * s + (dtv * xv) * Bn;
        float p = s * Cn;
        p += __shfl_xor(p, 1, 32);
        p += __shfl_xor(p, 2, 32);
        p += __shfl_xor(p, 4, 32);
        p += __shfl_xor(p, 8, 32);
        if (n == 0) {
            float yv = p + xv * Dp;
            int u = rev ? (L_ - 1 - l) : l;
            float zv = xz[(size_t)u * E_ + Di_ + d];
            y[(size_t)l * Di_ + d] = yv * (zv / (1.0f + __expf(-zv)));
        }
    }
}

// ---------------------------------------------------------------------------
// g[l,d] = y_fwd[l,d] + y_rev[L-1-l,d]  (bf16) — feeds the single out-proj.
// ---------------------------------------------------------------------------
__global__ void combine_kernel(const float* __restrict__ yf,
                               const float* __restrict__ yr,
                               __bf16* __restrict__ g) {
    int idx = blockIdx.x * blockDim.x + threadIdx.x;
    if (idx >= L_ * Di_) return;
    int l = idx / Di_;
    int d = idx % Di_;
    g[idx] = (__bf16)(yf[idx] + yr[(size_t)(L_ - 1 - l) * Di_ + d]);
}

// ---------------------------------------------------------------------------
// Host side
// ---------------------------------------------------------------------------
static inline int ceil_div(int a, int b) { return (a + b - 1) / b; }

template<int NT, int EPI>
static void launch_gemm(const __bf16* A, const __bf16* W, const float* bias,
                        float* C, int M, int N, int K, hipStream_t stream) {
    int tiles = (M / 16) * (N / (16 * NT));
    int threads = tiles * 32;
    int blocks = ceil_div(threads, 256);
    gemm_wmma_bf16<NT, EPI><<<blocks, 256, 0, stream>>>(A, W, bias, C, M, N, K);
}

extern "C" void kernel_launch(void* const* d_in, const int* in_sizes, int n_in,
                              void* d_out, int out_size, void* d_ws, size_t ws_size,
                              hipStream_t stream) {
    const float* h      = (const float*)d_in[0];
    const float* W_in   = (const float*)d_in[1];
    const float* W_out  = (const float*)d_in[2];
    const float* cw[2]  = { (const float*)d_in[3],  (const float*)d_in[10] };
    const float* cb[2]  = { (const float*)d_in[4],  (const float*)d_in[11] };
    const float* Wx[2]  = { (const float*)d_in[5],  (const float*)d_in[12] };
    const float* Wdt[2] = { (const float*)d_in[6],  (const float*)d_in[13] };
    const float* bdt[2] = { (const float*)d_in[7],  (const float*)d_in[14] };
    const float* Alog[2]= { (const float*)d_in[8],  (const float*)d_in[15] };
    const float* Dv[2]  = { (const float*)d_in[9],  (const float*)d_in[16] };

    // ---- workspace layout (all sizes multiples of 256B) ----
    char* w = (char*)d_ws;
    size_t off = 0;
    auto alloc = [&](size_t bytes) { char* p = w + off; off += bytes; return p; };

    __bf16* h_bf    = (__bf16*)alloc((size_t)L_ * Dm_ * 2);        //  4 MB
    __bf16* Win_bf  = (__bf16*)alloc((size_t)E_ * Dm_ * 2);        //  8 MB
    __bf16* Wout_bf = (__bf16*)alloc((size_t)Dm_ * Di_ * 2);       //  4 MB
    __bf16* Wx_bf[2], *Wdt_bf[2];
    float *xfbuf[2], *xdbl[2], *dtbuf[2], *ybuf[2];
    __bf16 *xb[2], *dtr[2];
    for (int s = 0; s < 2; ++s) Wx_bf[s]  = (__bf16*)alloc((size_t)G_ * Di_ * 2);
    for (int s = 0; s < 2; ++s) Wdt_bf[s] = (__bf16*)alloc((size_t)Di_ * R_ * 2);
    float* xzbuf = (float*)alloc((size_t)L_ * E_ * 4);             // 32 MB
    for (int s = 0; s < 2; ++s) xfbuf[s] = (float*)alloc((size_t)L_ * Di_ * 4);
    for (int s = 0; s < 2; ++s) xb[s]    = (__bf16*)alloc((size_t)L_ * Di_ * 2);
    for (int s = 0; s < 2; ++s) xdbl[s]  = (float*)alloc((size_t)L_ * G_ * 4);
    for (int s = 0; s < 2; ++s) dtr[s]   = (__bf16*)alloc((size_t)L_ * R_ * 2);
    for (int s = 0; s < 2; ++s) dtbuf[s] = (float*)alloc((size_t)L_ * Di_ * 4);
    for (int s = 0; s < 2; ++s) ybuf[s]  = (float*)alloc((size_t)L_ * Di_ * 4);
    __bf16* g_bf = (__bf16*)alloc((size_t)L_ * Di_ * 2);           //  8 MB
    (void)ws_size; // total ~180 MB

    auto cvt = [&](const float* src, __bf16* dst, int n) {
        cvt_f32_bf16_kernel<<<ceil_div(n, 256), 256, 0, stream>>>(src, dst, n);
    };

    // ---- stage bf16 copies of weights & activations ----
    cvt(h,     h_bf,    L_ * Dm_);
    cvt(W_in,  Win_bf,  E_ * Dm_);
    cvt(W_out, Wout_bf, Dm_ * Di_);
    for (int s = 0; s < 2; ++s) { cvt(Wx[s], Wx_bf[s], G_ * Di_);
                                  cvt(Wdt[s], Wdt_bf[s], Di_ * R_); }

    // ---- in-projection once (shared; reverse = flip of rows) ----
    launch_gemm<4, 0>(h_bf, Win_bf, nullptr, xzbuf, L_, E_, Dm_, stream);

    // ---- per-direction pipeline ----
    for (int s = 0; s < 2; ++s) {
        conv_silu_kernel<<<ceil_div(L_ * Di_, 256), 256, 0, stream>>>(
            xzbuf, cw[s], cb[s], xfbuf[s], xb[s], s);
        launch_gemm<2, 0>(xb[s], Wx_bf[s], nullptr, xdbl[s], L_, G_, Di_, stream);
        extract_dtr_kernel<<<ceil_div(L_ * R_, 256), 256, 0, stream>>>(
            xdbl[s], dtr[s]);
        launch_gemm<4, 1>(dtr[s], Wdt_bf[s], bdt[s], dtbuf[s], L_, Di_, R_, stream);
        scan_kernel<<<ceil_div(Di_ * 16, 256), 256, 0, stream>>>(
            dtbuf[s], xfbuf[s], xdbl[s], xzbuf, Alog[s], Dv[s], ybuf[s], s);
    }

    // ---- combine directions, single out-projection ----
    combine_kernel<<<ceil_div(L_ * Di_, 256), 256, 0, stream>>>(
        ybuf[0], ybuf[1], g_bf);
    launch_gemm<4, 0>(g_bf, Wout_bf, nullptr, (float*)d_out, L_, Dm_, Di_, stream);
}
